// LAS_1090921693202
// MI455X (gfx1250) — matrix operand order repaired and checked by reading, not compiled, not run on hardware
//
// LAS (Listen-Attend-Spell) forward for MI455X (gfx1250), bf16 WMMA everywhere.
//
// Roofline reasoning: all weights (<4 MB) are L2-resident (192 MB L2), HBM
// traffic is ~0.5 GB of activations vs ~23.3 TB/s -> the model is bound by
// small sequentially-dependent GEMMs. We therefore (a) run every matmul on
// v_wmma_f32_16x16x32_bf16, (b) keep LSTM hidden state in LDS and cell state
// in registers inside persistent 32-wave workgroups, (c) exploit the 320 KB
// WGP LDS to keep the whole speller working set on-chip, and (d) use the
// Tensor Data Mover to double-buffer x_t into LDS ahead of the recurrent
// compute (issue TDM for t+1, s_wait_tensorcnt + barrier per step).

#include <hip/hip_runtime.h>
#include <hip/hip_bf16.h>
#include <math.h>

typedef __bf16 bf16_t;
typedef __attribute__((ext_vector_type(16))) __bf16 v16bf;
typedef __attribute__((ext_vector_type(8)))  float  v8f;
typedef __attribute__((ext_vector_type(4)))  unsigned int v4u;
typedef __attribute__((ext_vector_type(8)))  int v8i;
typedef __attribute__((ext_vector_type(4)))  int v4i;

#define DEVI __device__ __forceinline__

DEVI v8f zero8() {
  v8f z;
#pragma unroll
  for (int i = 0; i < 8; ++i) z[i] = 0.f;
  return z;
}

DEVI float sigf(float x) { return 1.f / (1.f + __expf(-x)); }

// ---- WMMA fragment loaders (CDNA5 16x16x32 bf16 layouts, wave32) ----
// A (16xK=32): lanes 0-15 rows M=0..15 hold K 0..7 / 16..23 ; lanes 16-31 same
// rows, K 8..15 / 24..31 (2 halves per VGPR).
DEVI v16bf load_a_frag(const bf16_t* p, int ld) {
  int lane = threadIdx.x & 31;
  const bf16_t* r = p + (size_t)(lane & 15) * ld + ((lane >> 4) << 3);
  v16bf a;
#pragma unroll
  for (int i = 0; i < 8; ++i) { a[i] = r[i]; a[i + 8] = r[i + 16]; }
  return a;
}
// B (K=32 x N=16) sourced from row-major W(N,K): lane n holds K 0..7 / 16..23
// (lanes 0-15) or K 8..15 / 24..31 (lanes 16-31) of its column.
DEVI v16bf load_b_frag(const bf16_t* p, int ld) {
  int lane = threadIdx.x & 31;
  const bf16_t* r = p + (size_t)(lane & 15) * ld + ((lane >> 4) << 3);
  v16bf b;
#pragma unroll
  for (int i = 0; i < 8; ++i) { b[i] = r[i]; b[i + 8] = r[i + 16]; }
  return b;
}

DEVI v8f wmma_bf(v16bf a, v16bf b, v8f c) {
  return __builtin_amdgcn_wmma_f32_16x16x32_bf16(false, a, false, b,
                                                 (short)0, c, false, false);
}

// ---- Tensor Data Mover: 1-row tile copy global->LDS (ISA ch.8 D# layout) ----
// Copies nelem bf16 elements (nelem <= 32768) starting at gsrc into LDS at
// ldst. D# group0: count=1 | lds_addr | global_addr | type=2("image").
// Group1: data_size=1(2B), tensor_dim0=tile_dim0=nelem, dims1/2 unused.
DEVI void tdm_load_row(const bf16_t* gsrc, bf16_t* ldst, int nelem) {
  unsigned long long ga = (unsigned long long)(size_t)gsrc;
  unsigned int lds = (unsigned int)(size_t)ldst;  // low 32 bits = LDS offset
  v4u g0;
  g0[0] = 1u;                                          // count=1, user D#
  g0[1] = lds;                                         // lds_addr
  g0[2] = (unsigned int)(ga & 0xFFFFFFFFu);            // global_addr[31:0]
  g0[3] = (unsigned int)((ga >> 32) & 0x1FFFFFFu) | (2u << 30);  // [56:32]|type=2
  v8i g1;
  g1[0] = (1 << 16);                       // data_size=1 (2 bytes/elem)
  g1[1] = (nelem & 0xFFFF) << 16;          // tensor_dim0[15:0] (abar addr = 0)
  g1[2] = ((unsigned)nelem >> 16) | (1 << 16);  // tensor_dim0[31:16], tensor_dim1=1
  g1[3] = (nelem & 0xFFFF) << 16;          // tile_dim0 = nelem
  g1[4] = 0;                               // tile_dim1=0 (unused), tile_dim2=0
  g1[5] = nelem;                           // tensor_dim0_stride[31:0]
  g1[6] = 0;                               // stride[47:32]=0
  g1[7] = 0;
  v4i z4; z4[0] = 0; z4[1] = 0; z4[2] = 0; z4[3] = 0;
#if __clang_major__ >= 23
  v8i z8; z8[0]=0; z8[1]=0; z8[2]=0; z8[3]=0; z8[4]=0; z8[5]=0; z8[6]=0; z8[7]=0;
  __builtin_amdgcn_tensor_load_to_lds(g0, g1, z4, z4, z8, 0);
#else
  __builtin_amdgcn_tensor_load_to_lds(g0, g1, z4, z4, 0);
#endif
}

// Gate math for one 16x16 tile-group (i,f,g,o accumulators share (row,col)).
DEVI void lstm_gate(const v8f* acc, const float* bi, float* c, float* h) {
#pragma unroll
  for (int j = 0; j < 8; ++j) {
    float iv = sigf(acc[0][j] + bi[0]);
    float fv = sigf(acc[1][j] + bi[1]);
    float gv = tanhf(acc[2][j] + bi[2]);
    float ov = sigf(acc[3][j] + bi[3]);
    float cc = fv * c[j] + iv * gv;
    c[j] = cc;
    h[j] = ov * tanhf(cc);
  }
}

// =============================== repack kernels ===============================
__global__ void repack_kernel(const float* __restrict__ in, bf16_t* __restrict__ out,
                              int rows_in, int rows_out, int cols_in, int cols_out) {
  int idx = blockIdx.x * blockDim.x + threadIdx.x;
  int total = rows_out * cols_out;
  if (idx >= total) return;
  int r = idx / cols_out, c = idx - r * cols_out;
  float v = (r < rows_in && c < cols_in) ? in[(size_t)r * cols_in + c] : 0.f;
  out[idx] = (bf16_t)v;
}

// conv weight (O=128, C=256, K=2) -> Wc[o][k*256+c] so conv becomes a K=512 GEMM
__global__ void conv_repack_kernel(const float* __restrict__ in, bf16_t* __restrict__ out) {
  int idx = blockIdx.x * blockDim.x + threadIdx.x;
  if (idx >= 128 * 512) return;
  int o = idx >> 9, kc = idx & 511, k = kc >> 8, c = kc & 255;
  out[idx] = (bf16_t)in[(size_t)o * 512 + c * 2 + k];
}

// utter (B=256, T=1024, 40) f32 -> x0 (T, B, 64) bf16, zero-padded K 40..63
__global__ void utter_pack_kernel(const float* __restrict__ u, bf16_t* __restrict__ x0) {
  int idx = blockIdx.x * blockDim.x + threadIdx.x;
  if (idx >= 1024 * 256 * 64) return;
  int c = idx & 63, b = (idx >> 6) & 255, t = idx >> 14;
  x0[idx] = (bf16_t)((c < 40) ? u[((size_t)b * 1024 + t) * 40 + c] : 0.f);
}

// embedding gather: emb_bf (L=200, B=256, 256)
__global__ void embed_gather_kernel(const float* __restrict__ embed,
                                    const int* __restrict__ targets,
                                    bf16_t* __restrict__ out) {
  int idx = blockIdx.x * blockDim.x + threadIdx.x;
  if (idx >= 200 * 256 * 256) return;
  int e = idx & 255, b = (idx >> 8) & 255, l = idx >> 16;
  int tok = targets[b * 200 + l];
  out[idx] = (bf16_t)embed[(size_t)tok * 256 + e];
}

// =============================== BiLSTM layer ================================
// One persistent workgroup per direction (gridDim.x == 2), 1024 threads.
// LDS: h (256x128 bf16) + double-buffered x_t tiles filled by the TDM.
// z = x_t @ Wih^T + h @ Whh^T + b ; tasks = 16 M-tiles x 8 h-column-tiles.
__global__ void bilstm_kernel(const bf16_t* __restrict__ x, int T, int Dpad,
                              const bf16_t* __restrict__ wihf, const bf16_t* __restrict__ whhf,
                              const float* __restrict__ bf_,
                              const bf16_t* __restrict__ wihr, const bf16_t* __restrict__ whhr,
                              const float* __restrict__ br_,
                              bf16_t* __restrict__ out) {
  extern __shared__ char smem[];
  bf16_t* hll = (bf16_t*)smem;  // 256*128 bf16 = 64KB
  bf16_t* xb0 = (bf16_t*)(smem + 65536);
  const int xelem = 256 * Dpad;
  bf16_t* xb1 = xb0 + xelem;
  const int dir = blockIdx.x;
  const bf16_t* wih = dir ? wihr : wihf;
  const bf16_t* whh = dir ? whhr : whhf;
  const float* bias = dir ? br_ : bf_;
  const int tid = threadIdx.x, wave = tid >> 5, lane = tid & 31;
  const int lrow = (lane >> 4) << 3, lcol = lane & 15;

  for (int i = tid; i < 256 * 128; i += 1024) hll[i] = (bf16_t)0.f;

  // TDM: prefill x buffer 0 with the first timestep (wave 0 owns TENSORcnt)
  if (wave == 0) {
    int t0 = dir ? (T - 1) : 0;
    tdm_load_row(x + (size_t)t0 * xelem, xb0, xelem);
  }

  float creg[4][8];
  for (int s = 0; s < 4; ++s)
    for (int j = 0; j < 8; ++j) creg[s][j] = 0.f;

  int mt_[4], nt_[4];
  float bi[4][4];
  for (int s = 0; s < 4; ++s) {
    int task = wave + 32 * s;
    mt_[s] = task >> 3;
    nt_[s] = task & 7;
    int col = nt_[s] * 16 + lcol;
    for (int g = 0; g < 4; ++g) bi[s][g] = bias[g * 128 + col];
  }
  if (wave == 0) __builtin_amdgcn_s_wait_tensorcnt(0);
  __syncthreads();

  const int KTx = Dpad >> 5;
  for (int tt = 0; tt < T; ++tt) {
    int t = dir ? (T - 1 - tt) : tt;
    bf16_t* xcur = (tt & 1) ? xb1 : xb0;
    bf16_t* xnxt = (tt & 1) ? xb0 : xb1;
    if (wave == 0 && tt + 1 < T) {  // async DMA of x_{t+1} while we compute
      int tn = dir ? (t - 1) : (t + 1);
      tdm_load_row(x + (size_t)tn * xelem, xnxt, xelem);
    }
    float hreg[4][8];
#pragma unroll
    for (int s = 0; s < 4; ++s) {
      int r0 = mt_[s] << 4, nt = nt_[s];
      v8f acc[4];
#pragma unroll
      for (int g = 0; g < 4; ++g) acc[g] = zero8();
      for (int kt = 0; kt < KTx; ++kt) {  // input projection (x from LDS)
        v16bf a = load_a_frag(xcur + (size_t)r0 * Dpad + kt * 32, Dpad);
        v16bf wb[4];
#pragma unroll
        for (int g = 0; g < 4; ++g)
          wb[g] = load_b_frag(wih + (size_t)(g * 128 + nt * 16) * Dpad + kt * 32, Dpad);
#pragma unroll
        for (int g = 0; g < 4; ++g) acc[g] = wmma_bf(a, wb[g], acc[g]);
      }
#pragma unroll
      for (int kt = 0; kt < 4; ++kt) {  // recurrent projection (h from LDS)
        v16bf a = load_a_frag(hll + r0 * 128 + kt * 32, 128);
        v16bf wb[4];
#pragma unroll
        for (int g = 0; g < 4; ++g)
          wb[g] = load_b_frag(whh + (size_t)(g * 128 + nt * 16) * 128 + kt * 32, 128);
#pragma unroll
        for (int g = 0; g < 4; ++g) acc[g] = wmma_bf(a, wb[g], acc[g]);
      }
      lstm_gate(acc, bi[s], creg[s], hreg[s]);
    }
    __syncthreads();  // all reads of h(t-1) complete
    bf16_t* orow = out + (size_t)t * 256 * 256 + dir * 128;
#pragma unroll
    for (int s = 0; s < 4; ++s) {
      int r0 = mt_[s] << 4, col = nt_[s] * 16 + lcol;
#pragma unroll
      for (int j = 0; j < 8; ++j) {
        bf16_t hb = (bf16_t)hreg[s][j];
        hll[(r0 + lrow + j) * 128 + col] = hb;
        orow[(size_t)(r0 + lrow + j) * 256 + col] = hb;
      }
    }
    if (wave == 0) __builtin_amdgcn_s_wait_tensorcnt(0);  // x_{t+1} landed
    __syncthreads();  // h(t) + next x visible
  }
}

// ============================ conv k2s2 as GEMM ==============================
// out(t2,b,o) = sum_{k,c} x[2*t2+k, b, c] * Wc[o][k*256+c] + bias[o]
__global__ void conv_gemm_kernel(const bf16_t* __restrict__ x, const bf16_t* __restrict__ w,
                                 const float* __restrict__ bias, bf16_t* __restrict__ out,
                                 int T2) {
  const int B = 256, C = 256, N = 128;
  int lane = threadIdx.x & 31;
  int wave = blockIdx.x * (blockDim.x >> 5) + (threadIdx.x >> 5);
  int nwaves = gridDim.x * (blockDim.x >> 5);
  int mtiles = (T2 * B) >> 4;
  int total = mtiles * 8;
  for (int task = wave; task < total; task += nwaves) {
    int mt = task >> 3, nt = task & 7;
    int r0 = mt << 4;
    int t2 = r0 / B, b0 = r0 % B;  // B % 16 == 0 -> tile shares t2
    const bf16_t* x0 = x + (size_t)(2 * t2) * B * C + (size_t)b0 * C;
    const bf16_t* x1 = x0 + (size_t)B * C;
    __builtin_prefetch(x1 + (size_t)B * C, 0, 1);  // next t2 (global_prefetch_b8)
    v8f acc = zero8();
#pragma unroll
    for (int kt = 0; kt < 16; ++kt) {
      int kb = kt * 32;
      const bf16_t* ap = (kb < 256) ? (x0 + kb) : (x1 + kb - 256);
      acc = wmma_bf(load_a_frag(ap, C),
                    load_b_frag(w + (size_t)(nt * 16) * 512 + kb, 512), acc);
    }
    int col = nt * 16 + (lane & 15);
    float bc = bias[col];
    bf16_t* op = out + (size_t)r0 * N + col;
#pragma unroll
    for (int j = 0; j < 8; ++j) {
      int row = ((lane >> 4) << 3) + j;
      op[(size_t)row * N] = (bf16_t)(acc[j] + bc);
    }
  }
}

// ====================== key/value projection (f32 out) =======================
// A rows r = t*B + b of l3out (T8,B,256); out stored (B, T8, 128) f32.
__global__ void kv_kernel(const bf16_t* __restrict__ x, const bf16_t* __restrict__ w,
                          const float* __restrict__ bias, float* __restrict__ out) {
  const int B = 256, T8 = 128;
  int lane = threadIdx.x & 31;
  int wave = blockIdx.x * (blockDim.x >> 5) + (threadIdx.x >> 5);
  int nwaves = gridDim.x * (blockDim.x >> 5);
  int total = ((T8 * B) >> 4) * 8;
  for (int task = wave; task < total; task += nwaves) {
    int mt = task >> 3, nt = task & 7;
    int r0 = mt << 4;
    int t = r0 / B, b0 = r0 % B;
    v8f acc = zero8();
#pragma unroll
    for (int kt = 0; kt < 8; ++kt)
      acc = wmma_bf(load_a_frag(x + (size_t)r0 * 256 + kt * 32, 256),
                    load_b_frag(w + (size_t)(nt * 16) * 256 + kt * 32, 256), acc);
    int col = nt * 16 + (lane & 15);
    float bc = bias[col];
#pragma unroll
    for (int j = 0; j < 8; ++j) {
      int b = b0 + ((lane >> 4) << 3) + j;
      out[((size_t)b * T8 + t) * 128 + col] = acc[j] + bc;
    }
  }
}

// ================================= speller ===================================
// One persistent workgroup, 1024 threads, ~273KB dynamic LDS (CDNA5 320KB WGP).
__global__ void speller_kernel(
    const bf16_t* __restrict__ emb, const float* __restrict__ hk, const float* __restrict__ hv,
    const bf16_t* __restrict__ c0wih, const bf16_t* __restrict__ c0whh, const float* __restrict__ c0b,
    const bf16_t* __restrict__ c1wih, const bf16_t* __restrict__ c1whh, const float* __restrict__ c1b,
    const bf16_t* __restrict__ l1w, const float* __restrict__ l1b,
    const bf16_t* __restrict__ l2w, const float* __restrict__ l2b,
    const bf16_t* __restrict__ l3p, const float* __restrict__ l3b,
    const float* __restrict__ bng, const float* __restrict__ bnb,
    float* __restrict__ preds, float* __restrict__ attn) {
  extern __shared__ char smem[];
  bf16_t* sh0   = (bf16_t*)smem;                 // 64KB
  bf16_t* sh1   = (bf16_t*)(smem + 65536);       // 64KB
  bf16_t* ctxb  = (bf16_t*)(smem + 131072);      // 64KB
  bf16_t* attob = (bf16_t*)(smem + 196608);      // 64KB (att weights, reused as relu-out)
  float*  red   = (float*)(smem + 262144);       // 16KB BN partials
  float*  mv    = red + 4096;                    // mean / invstd

  const int tid = threadIdx.x, wave = tid >> 5, lane = tid & 31;
  const int lrow = (lane >> 4) << 3, lcol = lane & 15;

  for (int i = tid; i < 32768; i += 1024) {
    sh0[i] = (bf16_t)0.f; sh1[i] = (bf16_t)0.f; ctxb[i] = (bf16_t)0.f;
  }
  float c0reg[4][8], c1reg[4][8];
  for (int s = 0; s < 4; ++s)
    for (int j = 0; j < 8; ++j) { c0reg[s][j] = 0.f; c1reg[s][j] = 0.f; }

  int mt_[4], nt_[4];
  float c0bi[4][4], c1bi[4][4], lbias[4], bg_[4], bb_[4];
  for (int s = 0; s < 4; ++s) {
    int task = wave + 32 * s;
    mt_[s] = task >> 3; nt_[s] = task & 7;
    int col = nt_[s] * 16 + lcol;
    for (int g = 0; g < 4; ++g) { c0bi[s][g] = c0b[g * 128 + col]; c1bi[s][g] = c1b[g * 128 + col]; }
    lbias[s] = l1b[col] + l2b[col];
    bg_[s] = bng[col]; bb_[s] = bnb[col];
  }
  __syncthreads();

  for (int l = 0; l < 200; ++l) {  // l==0 is the warm-up call (pred discarded)
    const bf16_t* ybase = emb + (size_t)(l ? l - 1 : 0) * (256 * 256);

    // ---- cell 0: z0 = [y,ctx]@c0_Wih^T + sh0@c0_Whh^T ----
    float h0reg[4][8];
#pragma unroll
    for (int s = 0; s < 4; ++s) {
      int r0 = mt_[s] << 4, nt = nt_[s];
      v8f acc[4];
#pragma unroll
      for (int g = 0; g < 4; ++g) acc[g] = zero8();
#pragma unroll
      for (int kt = 0; kt < 8; ++kt) {  // y (K=256, Wih cols 0..255)
        v16bf a = load_a_frag(ybase + (size_t)r0 * 256 + kt * 32, 256);
        v16bf wb[4];
#pragma unroll
        for (int g = 0; g < 4; ++g)
          wb[g] = load_b_frag(c0wih + (size_t)(g * 128 + nt * 16) * 384 + kt * 32, 384);
#pragma unroll
        for (int g = 0; g < 4; ++g) acc[g] = wmma_bf(a, wb[g], acc[g]);
      }
#pragma unroll
      for (int kt = 0; kt < 4; ++kt) {  // ctx (K=128, Wih cols 256..383)
        v16bf a = load_a_frag(ctxb + r0 * 128 + kt * 32, 128);
        v16bf wb[4];
#pragma unroll
        for (int g = 0; g < 4; ++g)
          wb[g] = load_b_frag(c0wih + (size_t)(g * 128 + nt * 16) * 384 + 256 + kt * 32, 384);
#pragma unroll
        for (int g = 0; g < 4; ++g) acc[g] = wmma_bf(a, wb[g], acc[g]);
      }
#pragma unroll
      for (int kt = 0; kt < 4; ++kt) {  // recurrent sh0
        v16bf a = load_a_frag(sh0 + r0 * 128 + kt * 32, 128);
        v16bf wb[4];
#pragma unroll
        for (int g = 0; g < 4; ++g)
          wb[g] = load_b_frag(c0whh + (size_t)(g * 128 + nt * 16) * 128 + kt * 32, 128);
#pragma unroll
        for (int g = 0; g < 4; ++g) acc[g] = wmma_bf(a, wb[g], acc[g]);
      }
      lstm_gate(acc, c0bi[s], c0reg[s], h0reg[s]);
    }
    __syncthreads();
#pragma unroll
    for (int s = 0; s < 4; ++s) {
      int r0 = mt_[s] << 4, col = nt_[s] * 16 + lcol;
#pragma unroll
      for (int j = 0; j < 8; ++j) sh0[(r0 + lrow + j) * 128 + col] = (bf16_t)h0reg[s][j];
    }
    __syncthreads();

    // ---- cell 1: z1 = sh0@c1_Wih^T + sh1@c1_Whh^T ----
    float h1reg[4][8];
#pragma unroll
    for (int s = 0; s < 4; ++s) {
      int r0 = mt_[s] << 4, nt = nt_[s];
      v8f acc[4];
#pragma unroll
      for (int g = 0; g < 4; ++g) acc[g] = zero8();
#pragma unroll
      for (int kt = 0; kt < 4; ++kt) {
        v16bf a = load_a_frag(sh0 + r0 * 128 + kt * 32, 128);
        v16bf wb[4];
#pragma unroll
        for (int g = 0; g < 4; ++g)
          wb[g] = load_b_frag(c1wih + (size_t)(g * 128 + nt * 16) * 128 + kt * 32, 128);
#pragma unroll
        for (int g = 0; g < 4; ++g) acc[g] = wmma_bf(a, wb[g], acc[g]);
      }
#pragma unroll
      for (int kt = 0; kt < 4; ++kt) {
        v16bf a = load_a_frag(sh1 + r0 * 128 + kt * 32, 128);
        v16bf wb[4];
#pragma unroll
        for (int g = 0; g < 4; ++g)
          wb[g] = load_b_frag(c1whh + (size_t)(g * 128 + nt * 16) * 128 + kt * 32, 128);
#pragma unroll
        for (int g = 0; g < 4; ++g) acc[g] = wmma_bf(a, wb[g], acc[g]);
      }
      lstm_gate(acc, c1bi[s], c1reg[s], h1reg[s]);
    }
    __syncthreads();
#pragma unroll
    for (int s = 0; s < 4; ++s) {
      int r0 = mt_[s] << 4, col = nt_[s] * 16 + lcol;
#pragma unroll
      for (int j = 0; j < 8; ++j) sh1[(r0 + lrow + j) * 128 + col] = (bf16_t)h1reg[s][j];
    }
    __syncthreads();

    // ---- attention scores + softmax (4 lanes per batch row, wave32 shfl) ----
    {
      int b = tid >> 2, q = tid & 3;
      float sc[32];
      const float* hkb = hk + (size_t)b * (128 * 128);
#pragma unroll 4
      for (int jj = 0; jj < 32; ++jj) {
        const float* hkt = hkb + (q * 32 + jj) * 128;
        float s = 0.f;
        for (int o = 0; o < 128; ++o) s += hkt[o] * (float)sh1[b * 128 + o];
        sc[jj] = s;
      }
      float m = sc[0];
#pragma unroll
      for (int jj = 1; jj < 32; ++jj) m = fmaxf(m, sc[jj]);
      m = fmaxf(m, __shfl_xor(m, 1)); m = fmaxf(m, __shfl_xor(m, 2));
      float sum = 0.f;
#pragma unroll
      for (int jj = 0; jj < 32; ++jj) { sc[jj] = __expf(sc[jj] - m); sum += sc[jj]; }
      sum += __shfl_xor(sum, 1); sum += __shfl_xor(sum, 2);
      float inv = 1.f / sum;  // full mask -> renormalization is identity
#pragma unroll
      for (int jj = 0; jj < 32; ++jj) attob[b * 128 + q * 32 + jj] = (bf16_t)(sc[jj] * inv);
    }
    __syncthreads();
    if (l > 0 && tid < 128) attn[(size_t)(l - 1) * 128 + tid] = (float)attob[tid];

    // ---- context: ctx[b,o] = sum_t att[b,t]*hv[b,t,o] ----
#pragma unroll
    for (int z = 0; z < 32; ++z) {
      int idx = z * 1024 + tid;
      int b = idx >> 7, o = idx & 127;
      const float* hvb = hv + (size_t)b * (128 * 128) + o;
      float a = 0.f;
      for (int t = 0; t < 128; ++t) a += (float)attob[b * 128 + t] * hvb[t * 128];
      ctxb[idx] = (bf16_t)a;
    }
    __syncthreads();

    // ---- out = sh1@l1^T + ctx@l2^T + biases ; BN partial sums ----
    float outreg[4][8];
#pragma unroll
    for (int s = 0; s < 4; ++s) {
      int r0 = mt_[s] << 4, nt = nt_[s];
      v8f acc = zero8();
#pragma unroll
      for (int kt = 0; kt < 4; ++kt) {
        acc = wmma_bf(load_a_frag(sh1 + r0 * 128 + kt * 32, 128),
                      load_b_frag(l1w + (size_t)(nt * 16) * 128 + kt * 32, 128), acc);
        acc = wmma_bf(load_a_frag(ctxb + r0 * 128 + kt * 32, 128),
                      load_b_frag(l2w + (size_t)(nt * 16) * 128 + kt * 32, 128), acc);
      }
      float s1 = 0.f, s2 = 0.f;
#pragma unroll
      for (int j = 0; j < 8; ++j) {
        float v = acc[j] + lbias[s];
        outreg[s][j] = v; s1 += v; s2 += v * v;
      }
      s1 += __shfl_xor(s1, 16); s2 += __shfl_xor(s2, 16);  // fold M rows 0-7 with 8-15
      if (lane < 16) {
        red[mt_[s] * 128 + nt * 16 + lane] = s1;
        red[2048 + mt_[s] * 128 + nt * 16 + lane] = s2;
      }
    }
    __syncthreads();
    if (tid < 128) {  // per-column batch stats (biased var, B=256)
      float S = 0.f, Q = 0.f;
#pragma unroll
      for (int m2 = 0; m2 < 16; ++m2) { S += red[m2 * 128 + tid]; Q += red[2048 + m2 * 128 + tid]; }
      float mean = S * (1.f / 256.f);
      float var = Q * (1.f / 256.f) - mean * mean;
      mv[tid] = mean; mv[128 + tid] = rsqrtf(var + 1e-5f);
    }
    __syncthreads();
#pragma unroll
    for (int s = 0; s < 4; ++s) {  // normalize + relu -> attob (reused buffer)
      int r0 = mt_[s] << 4, col = nt_[s] * 16 + lcol;
      float mean = mv[col], istd = mv[128 + col];
#pragma unroll
      for (int j = 0; j < 8; ++j) {
        float v = bg_[s] * (outreg[s][j] - mean) * istd + bb_[s];
        attob[(r0 + lrow + j) * 128 + col] = (bf16_t)fmaxf(v, 0.f);
      }
    }
    __syncthreads();

    // ---- prediction: pred = relu_out @ l3^T (N padded 34->48) ----
    for (int pt = wave; pt < 48; pt += 32) {
      int mt = pt / 3, nt = pt % 3, r0 = mt << 4;
      v8f acc = zero8();
#pragma unroll
      for (int kt = 0; kt < 4; ++kt)
        acc = wmma_bf(load_a_frag(attob + r0 * 128 + kt * 32, 128),
                      load_b_frag(l3p + (size_t)(nt * 16) * 128 + kt * 32, 128), acc);
      int col = nt * 16 + lcol;
      if (l > 0 && col < 34) {
        float bias = l3b[col];
#pragma unroll
        for (int j = 0; j < 8; ++j)
          preds[(size_t)(r0 + lrow + j) * (199 * 34) + (size_t)(l - 1) * 34 + col] = acc[j] + bias;
      }
    }
    __syncthreads();  // protect attob/ctx for next step
  }
}

// ================================== host =====================================
extern "C" void kernel_launch(void* const* d_in, const int* in_sizes, int n_in,
                              void* d_out, int out_size, void* d_ws, size_t ws_size,
                              hipStream_t stream) {
  (void)in_sizes; (void)n_in; (void)out_size; (void)ws_size;
  const float* utter   = (const float*)d_in[0];
  const int*   targets = (const int*)d_in[1];
  const float* rWih[4][2]; const float* rWhh[4][2]; const float* rB[4][2];
  for (int l = 0; l < 4; ++l) {
    int base = 2 + l * 6;
    rWih[l][0] = (const float*)d_in[base + 0];
    rWhh[l][0] = (const float*)d_in[base + 1];
    rB  [l][0] = (const float*)d_in[base + 2];
    rWih[l][1] = (const float*)d_in[base + 3];
    rWhh[l][1] = (const float*)d_in[base + 4];
    rB  [l][1] = (const float*)d_in[base + 5];
  }
  const float* conv_w[3]; const float* conv_b[3];
  for (int i = 0; i < 3; ++i) { conv_w[i] = (const float*)d_in[26 + 2 * i]; conv_b[i] = (const float*)d_in[27 + 2 * i]; }
  const float* k_w = (const float*)d_in[32]; const float* k_b = (const float*)d_in[33];
  const float* v_w = (const float*)d_in[34]; const float* v_b = (const float*)d_in[35];
  const float* embed = (const float*)d_in[36];
  const float* c0_Wih = (const float*)d_in[37]; const float* c0_Whh = (const float*)d_in[38]; const float* c0_b = (const float*)d_in[39];
  const float* c1_Wih = (const float*)d_in[40]; const float* c1_Whh = (const float*)d_in[41]; const float* c1_b = (const float*)d_in[42];
  const float* l1_w = (const float*)d_in[43]; const float* l1_b = (const float*)d_in[44];
  const float* l2_w = (const float*)d_in[45]; const float* l2_b = (const float*)d_in[46];
  const float* l3_w = (const float*)d_in[47]; const float* l3_b = (const float*)d_in[48];
  const float* bn_g = (const float*)d_in[49]; const float* bn_b = (const float*)d_in[50];

  char* wsb = (char*)d_ws; size_t off = 0;
  auto alloc = [&](size_t bytes) -> void* {
    void* P = wsb + off; off = (off + bytes + 255) & ~(size_t)255; return P;
  };
  // bf16 weights (padded where needed)
  bf16_t* whh_[4][2]; bf16_t* wih_[4][2];
  wih_[0][0] = (bf16_t*)alloc(512 * 64 * 2); wih_[0][1] = (bf16_t*)alloc(512 * 64 * 2);
  for (int l = 1; l < 4; ++l) { wih_[l][0] = (bf16_t*)alloc(512 * 128 * 2); wih_[l][1] = (bf16_t*)alloc(512 * 128 * 2); }
  for (int l = 0; l < 4; ++l) { whh_[l][0] = (bf16_t*)alloc(512 * 128 * 2); whh_[l][1] = (bf16_t*)alloc(512 * 128 * 2); }
  bf16_t* cwb[3]; for (int i = 0; i < 3; ++i) cwb[i] = (bf16_t*)alloc(128 * 512 * 2);
  bf16_t* kwb = (bf16_t*)alloc(128 * 256 * 2);
  bf16_t* vwb = (bf16_t*)alloc(128 * 256 * 2);
  bf16_t* c0wihb = (bf16_t*)alloc(512 * 384 * 2);
  bf16_t* c0whhb = (bf16_t*)alloc(512 * 128 * 2);
  bf16_t* c1wihb = (bf16_t*)alloc(512 * 128 * 2);
  bf16_t* c1whhb = (bf16_t*)alloc(512 * 128 * 2);
  bf16_t* l1wb = (bf16_t*)alloc(128 * 128 * 2);
  bf16_t* l2wb = (bf16_t*)alloc(128 * 128 * 2);
  bf16_t* l3pb = (bf16_t*)alloc(48 * 128 * 2);
  // bf16 activations
  bf16_t* x0    = (bf16_t*)alloc((size_t)1024 * 256 * 64 * 2);
  bf16_t* l0out = (bf16_t*)alloc((size_t)1024 * 256 * 256 * 2);
  bf16_t* c1o   = (bf16_t*)alloc((size_t)512 * 256 * 128 * 2);
  bf16_t* l1out = (bf16_t*)alloc((size_t)512 * 256 * 256 * 2);
  bf16_t* c2o   = (bf16_t*)alloc((size_t)256 * 256 * 128 * 2);
  bf16_t* l2out = (bf16_t*)alloc((size_t)256 * 256 * 256 * 2);
  bf16_t* c3o   = (bf16_t*)alloc((size_t)128 * 256 * 128 * 2);
  bf16_t* l3out = (bf16_t*)alloc((size_t)128 * 256 * 256 * 2);
  float*  hkb   = (float*)alloc((size_t)256 * 128 * 128 * 4);
  float*  hvb   = (float*)alloc((size_t)256 * 128 * 128 * 4);
  bf16_t* embb  = (bf16_t*)alloc((size_t)200 * 256 * 256 * 2);

  auto rp = [&](const float* src, bf16_t* dst, int ri, int ro, int ci, int co) {
    int total = ro * co;
    repack_kernel<<<(total + 255) / 256, 256, 0, stream>>>(src, dst, ri, ro, ci, co);
  };
  // weight repacks
  rp(rWih[0][0], wih_[0][0], 512, 512, 40, 64);
  rp(rWih[0][1], wih_[0][1], 512, 512, 40, 64);
  for (int l = 1; l < 4; ++l) {
    rp(rWih[l][0], wih_[l][0], 512, 512, 128, 128);
    rp(rWih[l][1], wih_[l][1], 512, 512, 128, 128);
  }
  for (int l = 0; l < 4; ++l) {
    rp(rWhh[l][0], whh_[l][0], 512, 512, 128, 128);
    rp(rWhh[l][1], whh_[l][1], 512, 512, 128, 128);
  }
  for (int i = 0; i < 3; ++i)
    conv_repack_kernel<<<(128 * 512 + 255) / 256, 256, 0, stream>>>(conv_w[i], cwb[i]);
  rp(k_w, kwb, 128, 128, 256, 256);
  rp(v_w, vwb, 128, 128, 256, 256);
  rp(c0_Wih, c0wihb, 512, 512, 384, 384);
  rp(c0_Whh, c0whhb, 512, 512, 128, 128);
  rp(c1_Wih, c1wihb, 512, 512, 128, 128);
  rp(c1_Whh, c1whhb, 512, 512, 128, 128);
  rp(l1_w, l1wb, 128, 128, 128, 128);
  rp(l2_w, l2wb, 128, 128, 128, 128);
  rp(l3_w, l3pb, 34, 48, 128, 128);
  utter_pack_kernel<<<(1024 * 256 * 64 + 255) / 256, 256, 0, stream>>>(utter, x0);
  embed_gather_kernel<<<(200 * 256 * 256 + 255) / 256, 256, 0, stream>>>(embed, targets, embb);

  // ---- listener pipeline (LDS = h 64KB + 2 x_t buffers for TDM) ----
  bilstm_kernel<<<2, 1024, 65536 + 2 * 256 * 64 * 2, stream>>>(x0, 1024, 64,
      wih_[0][0], whh_[0][0], rB[0][0], wih_[0][1], whh_[0][1], rB[0][1], l0out);
  conv_gemm_kernel<<<512, 256, 0, stream>>>(l0out, cwb[0], conv_b[0], c1o, 512);
  bilstm_kernel<<<2, 1024, 65536 + 2 * 256 * 128 * 2, stream>>>(c1o, 512, 128,
      wih_[1][0], whh_[1][0], rB[1][0], wih_[1][1], whh_[1][1], rB[1][1], l1out);
  conv_gemm_kernel<<<512, 256, 0, stream>>>(l1out, cwb[1], conv_b[1], c2o, 256);
  bilstm_kernel<<<2, 1024, 65536 + 2 * 256 * 128 * 2, stream>>>(c2o, 256, 128,
      wih_[2][0], whh_[2][0], rB[2][0], wih_[2][1], whh_[2][1], rB[2][1], l2out);
  conv_gemm_kernel<<<512, 256, 0, stream>>>(l2out, cwb[2], conv_b[2], c3o, 128);
  bilstm_kernel<<<2, 1024, 65536 + 2 * 256 * 128 * 2, stream>>>(c3o, 128, 128,
      wih_[3][0], whh_[3][0], rB[3][0], wih_[3][1], whh_[3][1], rB[3][1], l3out);
  kv_kernel<<<256, 256, 0, stream>>>(l3out, kwb, k_b, hkb);
  kv_kernel<<<256, 256, 0, stream>>>(l3out, vwb, v_b, hvb);

  // ---- speller ----
  float* preds = (float*)d_out;
  float* attn = preds + (size_t)256 * 199 * 34;
  const unsigned SPELLER_LDS = 262144 + 16384 + 1024;  // 273KB < 320KB WGP LDS
  speller_kernel<<<1, 1024, SPELLER_LDS, stream>>>(
      embb, hkb, hvb, c0wihb, c0whhb, c0_b, c1wihb, c1whhb, c1_b,
      l1wb, l1_b, l2wb, l2_b, l3pb, l3_b, bn_g, bn_b, preds, attn);
}
